// TransformerLayer_44581760532724
// MI455X (gfx1250) — matrix operand order repaired and checked by reading, not compiled, not run on hardware
//
#include <hip/hip_runtime.h>
#include <hip/hip_bf16.h>
#include <stdint.h>

#define D_MODEL 1024
#define SEQ     2048
#define BATCH   2
#define HEADS   16
#define HD      64
#define FF_DIM  4096
#define EPSV    1e-5f

typedef _Float16 h16;
typedef __attribute__((ext_vector_type(16))) _Float16 v16h;
typedef __attribute__((ext_vector_type(8)))  _Float16 v8h;
typedef __attribute__((ext_vector_type(8)))  float    v8f;
typedef __attribute__((ext_vector_type(4)))  int      v4i;

// CDNA5 async global->LDS copy (ASYNCcnt path), guarded so compile can't break.
#if defined(__HIP_DEVICE_COMPILE__) && defined(__gfx1250__) && \
    __has_builtin(__builtin_amdgcn_global_load_async_to_lds_b128)
#define USE_ASYNC_LDS 1
#else
#define USE_ASYNC_LDS 0
#endif

#if USE_ASYNC_LDS
#define ASYNC_CP16(dst, src)                                                   \
  __builtin_amdgcn_global_load_async_to_lds_b128(                              \
      (__attribute__((address_space(1))) v4i*)(src),                           \
      (__attribute__((address_space(3))) v4i*)(dst), 0, 0)
#define WAIT_ASYNC() asm volatile("s_wait_asynccnt 0x0" ::: "memory")
#endif

__device__ __forceinline__ v16h cat16(v8h lo, v8h hi) {
  return __builtin_shufflevector(lo, hi, 0,1,2,3,4,5,6,7,8,9,10,11,12,13,14,15);
}
__device__ __forceinline__ v16h ld16(const h16* p) {
  v8h lo = *(const v8h*)p;
  v8h hi = *(const v8h*)(p + 8);
  return cat16(lo, hi);
}
__device__ __forceinline__ v8f wmma_f16(v16h a, v16h b, v8f c) {
  return __builtin_amdgcn_wmma_f32_16x16x32_f16(false, a, false, b, (short)0, c, false, false);
}

// ---------------------------------------------------------------- converts
__global__ void cvt_f32_f16(const float* __restrict__ in, h16* __restrict__ out, int n) {
  int i = blockIdx.x * blockDim.x + threadIdx.x;
  if (i < n) out[i] = (h16)in[i];
}

// W is K x N (row-major f32)  ->  Wt is N x K (row-major f16)
__global__ void cvt_transpose_f32_f16(const float* __restrict__ W, h16* __restrict__ Wt,
                                      int K, int N) {
  int i = blockIdx.x * 256 + threadIdx.x;
  if (i >= K * N) return;
  int k = i % K, n = i / K;                 // i indexes Wt row-major
  Wt[i] = (h16)W[(size_t)k * N + n];
}

// ---------------------------------------------------------------- GEMM
// C[M,N] = A[M,K] (f16 row-major) @ B[K,N] + bias[N], with B given PRE-TRANSPOSED
// as Bt[N,K] (f16 row-major). Block: 256 threads = 8 waves, block tile 128x128,
// wave tile 16x128 (8 accumulators), K-step 32, double-buffered LDS, 1 barrier/step.
__global__ __launch_bounds__(256)
void gemm16(const h16* __restrict__ A, const h16* __restrict__ Bt,
            const float* __restrict__ bias, float* __restrict__ Cf,
            h16* __restrict__ Ch, int M, int N, int K, int relu)
{
  __shared__ __align__(16) h16 As[2][128][40];   // [buf][m][k] (+8 pad, 80B stride)
  __shared__ __align__(16) h16 Bs[2][128][40];   // [buf][n][k]

  const int tid  = threadIdx.x;
  const int lane = tid & 31, wave = tid >> 5;
  const int hi = lane >> 4, lm = lane & 15;
  const int kb = hi * 8;            // A-operand K base (ISA 16-bit A layout)
  const int klo = hi * 16;          // unused; B-operand K base is kb, same k order as A
  const int m0 = blockIdx.y * 128, n0 = blockIdx.x * 128;

  // tile loader: 128 rows x 32 k, 2x16B per thread
  const int ldr = tid >> 1;                 // 0..127
  const int lc  = (tid & 1) * 16;           // 0 or 16
  const h16* gA = A  + (size_t)(m0 + ldr) * K + lc;
  const h16* gB = Bt + (size_t)(n0 + ldr) * K + lc;

  v8f acc[8] = {};

  // preload all 8 B-operands into distinct regs, then run the 8-WMMA chain
  auto compute = [&](int bsel) {
    v16h a = cat16(*(const v8h*)&As[bsel][wave * 16 + lm][kb],
                   *(const v8h*)&As[bsel][wave * 16 + lm][16 + kb]);
    v16h b[8];
#pragma unroll
    for (int nt = 0; nt < 8; ++nt)
      b[nt] = cat16(*(const v8h*)&Bs[bsel][nt * 16 + lm][kb],
                    *(const v8h*)&Bs[bsel][nt * 16 + lm][16 + kb]);
#pragma unroll
    for (int nt = 0; nt < 8; ++nt)
      acc[nt] = wmma_f16(a, b[nt], acc[nt]);
  };

#if USE_ASYNC_LDS
  // prologue: k0 = 0 into buffer 0 (async, tracked by ASYNCcnt)
  ASYNC_CP16(&As[0][ldr][lc],     gA);
  ASYNC_CP16(&As[0][ldr][lc + 8], gA + 8);
  ASYNC_CP16(&Bs[0][ldr][lc],     gB);
  ASYNC_CP16(&Bs[0][ldr][lc + 8], gB + 8);
  WAIT_ASYNC();
  __syncthreads();
  int buf = 0;
  const int steps = K >> 5;
  for (int s = 0; s < steps - 1; ++s) {      // branch-free body: prefetch + compute
    const h16* nA = gA + (s + 1) * 32;
    const h16* nB = gB + (s + 1) * 32;
    ASYNC_CP16(&As[buf ^ 1][ldr][lc],     nA);
    ASYNC_CP16(&As[buf ^ 1][ldr][lc + 8], nA + 8);
    ASYNC_CP16(&Bs[buf ^ 1][ldr][lc],     nB);
    ASYNC_CP16(&Bs[buf ^ 1][ldr][lc + 8], nB + 8);
    compute(buf);
    WAIT_ASYNC();
    __syncthreads();
    buf ^= 1;
  }
  compute(buf);                              // peeled last step, no prefetch
#else
  for (int k0 = 0; k0 < K; k0 += 32) {
    v8h a0 = *(const v8h*)(gA + k0);
    v8h a1 = *(const v8h*)(gA + k0 + 8);
    v8h b0 = *(const v8h*)(gB + k0);
    v8h b1 = *(const v8h*)(gB + k0 + 8);
    *(v8h*)&As[0][ldr][lc]     = a0;
    *(v8h*)&As[0][ldr][lc + 8] = a1;
    *(v8h*)&Bs[0][ldr][lc]     = b0;
    *(v8h*)&Bs[0][ldr][lc + 8] = b1;
    __syncthreads();
    compute(0);
    __syncthreads();
  }
#endif

#pragma unroll
  for (int nt = 0; nt < 8; ++nt) {
#pragma unroll
    for (int r = 0; r < 8; ++r) {
      const int row = m0 + wave * 16 + r + hi * 8;   // C/D: VGPR r <-> M = r + 8*hi
      const int col = n0 + nt * 16 + lm;
      float v = acc[nt][r] + bias[col];
      if (relu) v = fmaxf(v, 0.0f);
      if (Cf) Cf[(size_t)row * N + col] = v;
      if (Ch) Ch[(size_t)row * N + col] = (h16)v;
    }
  }
}

// ---------------------------------------------------------------- V transpose: [b,l,h,d] -> [b,h,d,l]
__global__ void vt_repack(const h16* __restrict__ vsrc, h16* __restrict__ vt) {
  int i = blockIdx.x * 256 + threadIdx.x;
  const int total = BATCH * HEADS * HD * SEQ;
  if (i >= total) return;
  int l  = i & (SEQ - 1);
  int t  = i >> 11;            // / SEQ
  int d  = t & (HD - 1);
  int bh = t >> 6;             // / HD
  int h  = bh & (HEADS - 1);
  int b  = bh >> 4;
  vt[i] = vsrc[(size_t)(b * SEQ + l) * D_MODEL + h * HD + d];
}

// ---------------------------------------------------------------- flash attention
// one wave per (b, h, 16-row q tile); 256 threads = 8 tiles per block.
__global__ __launch_bounds__(256)
void attn_kernel(const h16* __restrict__ q16, const h16* __restrict__ k16,
                 const h16* __restrict__ vt16, h16* __restrict__ msg16)
{
  __shared__ __align__(16) h16 Pls[8][16][32];  // wave-private P bounce buffer

  const int tid = threadIdx.x, lane = tid & 31, wave = tid >> 5;
  const int hi = lane >> 4, lm = lane & 15, kb = hi * 8;
  const int idx = blockIdx.x * 8 + wave;        // 0 .. B*H*(SEQ/16)-1
  const int qt = idx & (SEQ / 16 - 1);
  const int bh = idx >> 7;
  const int hh = bh & (HEADS - 1), bb = bh >> 4;
  h16 (*P)[32] = Pls[wave];

  // Q tile as two A-operands (head dims 0..31, 32..63)
  const size_t qoff = (size_t)(bb * SEQ + qt * 16 + lm) * D_MODEL + hh * HD;
  v16h a0 = cat16(*(const v8h*)(q16 + qoff + hi * 16),      *(const v8h*)(q16 + qoff + hi * 16 + 8));
  v16h a1 = cat16(*(const v8h*)(q16 + qoff + 32 + hi * 16), *(const v8h*)(q16 + qoff + 40 + hi * 16));

  v8f o[4] = {};
  float mrow[8], lrow[8];
#pragma unroll
  for (int r = 0; r < 8; ++r) { mrow[r] = -3.0e30f; lrow[r] = 0.0f; }

  const int klo = hi * 16;
  for (int k0 = 0; k0 < SEQ; k0 += 32) {
    // scores: S = Q @ K^T ; B-operand lane = key, K = head-dims (contiguous rows of k16)
    const size_t kr0 = (size_t)(bb * SEQ + k0 + lm) * D_MODEL + hh * HD;
    const size_t kr1 = kr0 + (size_t)16 * D_MODEL;
    v16h b00 = ld16(k16 + kr0 + klo);
    v16h b01 = ld16(k16 + kr0 + 32 + klo);
    v16h b10 = ld16(k16 + kr1 + klo);
    v16h b11 = ld16(k16 + kr1 + 32 + klo);
    v8f s0 = {}, s1 = {};
    s0 = wmma_f16(a0, b00, s0); s0 = wmma_f16(a1, b01, s0);
    s1 = wmma_f16(a0, b10, s1); s1 = wmma_f16(a1, b11, s1);

    // online softmax (row stats replicated across each 16-lane half)
#pragma unroll
    for (int r = 0; r < 8; ++r) {
      float x0 = s0[r] * 0.125f, x1 = s1[r] * 0.125f;   // scale = 1/sqrt(64)
      float bmax = fmaxf(x0, x1);
#pragma unroll
      for (int msk = 1; msk < 16; msk <<= 1) bmax = fmaxf(bmax, __shfl_xor(bmax, msk, 32));
      float nm = fmaxf(mrow[r], bmax);
      float f  = __expf(mrow[r] - nm);
      float p0 = __expf(x0 - nm), p1 = __expf(x1 - nm);
      float rs = p0 + p1;
#pragma unroll
      for (int msk = 1; msk < 16; msk <<= 1) rs += __shfl_xor(rs, msk, 32);
      lrow[r] = lrow[r] * f + rs;
      mrow[r] = nm;
#pragma unroll
      for (int nt = 0; nt < 4; ++nt) o[nt][r] = o[nt][r] * f;
      P[hi * 8 + r][lm]      = (h16)p0;
      P[hi * 8 + r][16 + lm] = (h16)p1;
    }

    // P (C-layout) -> A-operand via wave-private LDS bounce (DS in-order within wave)
    v16h pa = cat16(*(const v8h*)&P[lm][klo], *(const v8h*)&P[lm][klo + 8]);

    // O += P @ V : B-operand lane = head-dim, K = keys (contiguous in transposed Vt)
    const size_t vb = ((size_t)bh * HD) * SEQ + (size_t)k0 + klo;
    v16h bv[4];
#pragma unroll
    for (int nt = 0; nt < 4; ++nt)
      bv[nt] = ld16(vt16 + vb + (size_t)(nt * 16 + lm) * SEQ);
#pragma unroll
    for (int nt = 0; nt < 4; ++nt)
      o[nt] = wmma_f16(pa, bv[nt], o[nt]);
  }

#pragma unroll
  for (int nt = 0; nt < 4; ++nt) {
#pragma unroll
    for (int r = 0; r < 8; ++r) {
      const int row = qt * 16 + r + hi * 8;
      const float val = o[nt][r] / lrow[r];
      msg16[(size_t)(bb * SEQ + row) * D_MODEL + hh * HD + nt * 16 + lm] = (h16)val;
    }
  }
}

// ---------------------------------------------------------------- residual + layernorm
__global__ __launch_bounds__(256)
void ln_kernel(const float* __restrict__ X, const float* __restrict__ Y,
               const float* __restrict__ g, const float* __restrict__ be,
               float* __restrict__ out32, h16* __restrict__ out16)
{
  __shared__ float red[256];
  const int row = blockIdx.x, tid = threadIdx.x;
  const size_t base = (size_t)row * D_MODEL;
  float v[4]; float s = 0.f;
#pragma unroll
  for (int i = 0; i < 4; ++i) { int c = tid + i * 256; v[i] = X[base + c] + Y[base + c]; s += v[i]; }
  red[tid] = s; __syncthreads();
  for (int st = 128; st > 0; st >>= 1) { if (tid < st) red[tid] += red[tid + st]; __syncthreads(); }
  const float mu = red[0] * (1.0f / D_MODEL);
  __syncthreads();
  float s2 = 0.f;
#pragma unroll
  for (int i = 0; i < 4; ++i) { float d = v[i] - mu; s2 += d * d; }
  red[tid] = s2; __syncthreads();
  for (int st = 128; st > 0; st >>= 1) { if (tid < st) red[tid] += red[tid + st]; __syncthreads(); }
  const float rst = rsqrtf(red[0] * (1.0f / D_MODEL) + EPSV);
#pragma unroll
  for (int i = 0; i < 4; ++i) {
    int c = tid + i * 256;
    float y = (v[i] - mu) * rst * g[c] + be[c];
    out32[base + c] = y;
    if (out16) out16[base + c] = (h16)y;
  }
}

// ---------------------------------------------------------------- launcher
extern "C" void kernel_launch(void* const* d_in, const int* in_sizes, int n_in,
                              void* d_out, int out_size, void* d_ws, size_t ws_size,
                              hipStream_t stream)
{
  (void)in_sizes; (void)n_in; (void)out_size; (void)ws_size;
  const float* x   = (const float*)d_in[0];
  const float* Wq  = (const float*)d_in[1];
  const float* bq  = (const float*)d_in[2];
  const float* Wk  = (const float*)d_in[3];
  const float* bk  = (const float*)d_in[4];
  const float* Wv  = (const float*)d_in[5];
  const float* bv  = (const float*)d_in[6];
  const float* Wm  = (const float*)d_in[7];
  const float* bm  = (const float*)d_in[8];
  const float* W1  = (const float*)d_in[9];
  const float* b1  = (const float*)d_in[10];
  const float* W2  = (const float*)d_in[11];
  const float* b2  = (const float*)d_in[12];
  const float* g1  = (const float*)d_in[13];
  const float* be1 = (const float*)d_in[14];
  const float* g2  = (const float*)d_in[15];
  const float* be2 = (const float*)d_in[16];
  float* out = (float*)d_out;

  const int M = BATCH * SEQ;  // 4096
  char* ws = (char*)d_ws;
  size_t off = 0;
  auto alloc = [&](size_t bytes) -> char* {
    char* p = ws + off; off += (bytes + 255) & ~(size_t)255; return p;
  };
  h16* x16   = (h16*)alloc((size_t)M * D_MODEL * 2);
  h16* wq16  = (h16*)alloc((size_t)D_MODEL * D_MODEL * 2);   // transposed
  h16* wk16  = (h16*)alloc((size_t)D_MODEL * D_MODEL * 2);   // transposed
  h16* wv16  = (h16*)alloc((size_t)D_MODEL * D_MODEL * 2);   // transposed
  h16* wm16  = (h16*)alloc((size_t)D_MODEL * D_MODEL * 2);   // transposed
  h16* w116  = (h16*)alloc((size_t)D_MODEL * FF_DIM * 2);    // transposed: [FF, D]
  h16* w216  = (h16*)alloc((size_t)FF_DIM * D_MODEL * 2);    // transposed: [D, FF]
  h16* q16   = (h16*)alloc((size_t)M * D_MODEL * 2);
  h16* k16   = (h16*)alloc((size_t)M * D_MODEL * 2);
  h16* v16p  = (h16*)alloc((size_t)M * D_MODEL * 2);
  h16* vt16  = (h16*)alloc((size_t)BATCH * HEADS * HD * SEQ * 2);
  h16* msg16 = (h16*)alloc((size_t)M * D_MODEL * 2);
  float* mm32 = (float*)alloc((size_t)M * D_MODEL * 4);
  float* h32  = (float*)alloc((size_t)M * D_MODEL * 4);
  h16* h16b  = (h16*)alloc((size_t)M * D_MODEL * 2);
  h16* ff16  = q16;     // reuse q/k/v/vt block (32MB) after attention
  float* ff2 = mm32;    // reuse mm32 after ln1 consumed it

  dim3 blk(256);
  cvt_f32_f16<<<dim3((M * D_MODEL + 255) / 256), blk, 0, stream>>>(x, x16, M * D_MODEL);
  auto cvtT = [&](const float* w, h16* wt, int K, int N) {
    cvt_transpose_f32_f16<<<dim3((K * N + 255) / 256), blk, 0, stream>>>(w, wt, K, N);
  };
  cvtT(Wq, wq16, D_MODEL, D_MODEL);
  cvtT(Wk, wk16, D_MODEL, D_MODEL);
  cvtT(Wv, wv16, D_MODEL, D_MODEL);
  cvtT(Wm, wm16, D_MODEL, D_MODEL);
  cvtT(W1, w116, D_MODEL, FF_DIM);
  cvtT(W2, w216, FF_DIM, D_MODEL);

  dim3 gProj(D_MODEL / 128, M / 128);   // (8, 32)
  gemm16<<<gProj, blk, 0, stream>>>(x16, wq16, bq, nullptr, q16,  M, D_MODEL, D_MODEL, 0);
  gemm16<<<gProj, blk, 0, stream>>>(x16, wk16, bk, nullptr, k16,  M, D_MODEL, D_MODEL, 0);
  gemm16<<<gProj, blk, 0, stream>>>(x16, wv16, bv, nullptr, v16p, M, D_MODEL, D_MODEL, 0);

  vt_repack<<<dim3((BATCH * HEADS * HD * SEQ + 255) / 256), blk, 0, stream>>>(v16p, vt16);
  attn_kernel<<<dim3(BATCH * HEADS * (SEQ / 16) / 8), blk, 0, stream>>>(q16, k16, vt16, msg16);

  gemm16<<<gProj, blk, 0, stream>>>(msg16, wm16, bm, mm32, nullptr, M, D_MODEL, D_MODEL, 0);
  ln_kernel<<<dim3(M), blk, 0, stream>>>(x, mm32, g1, be1, h32, h16b);

  dim3 gF1(FF_DIM / 128, M / 128);      // (32, 32)
  gemm16<<<gF1, blk, 0, stream>>>(h16b, w116, b1, nullptr, ff16, M, FF_DIM, D_MODEL, 1);
  gemm16<<<gProj, blk, 0, stream>>>(ff16, w216, b2, ff2, nullptr, M, D_MODEL, FF_DIM, 0);
  ln_kernel<<<dim3(M), blk, 0, stream>>>(h32, ff2, g2, be2, out, nullptr);
}
